// GraphReasoning_76261439307876
// MI455X (gfx1250) — matrix operand order repaired and checked by reading, not compiled, not run on hardware
//
#include <hip/hip_runtime.h>
#include <hip/hip_bf16.h>

// ---------------------------------------------------------------- types
typedef __attribute__((ext_vector_type(16))) __bf16         v16bf;
typedef __attribute__((ext_vector_type(8)))  float          v8f;
typedef __attribute__((ext_vector_type(8)))  unsigned short u16x8;

union BFVec { v16bf bf; u16x8 h[2]; };

// hardware f32 -> bf16 (v_cvt_bf16_f32 / pairs into v_cvt_pk_bf16_f32)
__device__ __forceinline__ unsigned short f2bf(float f) {
  __bf16 h = (__bf16)f;
  return __builtin_bit_cast(unsigned short, h);
}

__device__ __forceinline__ v8f v8f_zero() {
  v8f z;
#pragma unroll
  for (int i = 0; i < 8; ++i) z[i] = 0.f;
  return z;
}

// WMMA A-tile (16x32 bf16) from row-major bf16 matrix [rows x ldk] in global.
// Lane L: M = L%16; elements 0..7 -> K = base+0..7, 8..15 -> K = base+16..23,
// base = (L>=16)*8.  Two 16-byte loads.
__device__ __forceinline__ v16bf load_a(const unsigned short* __restrict__ W,
                                        int row0, int k0, int ldk, int lane) {
  const unsigned short* p = W + (size_t)(row0 + (lane & 15)) * ldk + k0 + ((lane >> 4) << 3);
  BFVec r;
  r.h[0] = *(const u16x8*)(p);
  r.h[1] = *(const u16x8*)(p + 16);
  return r.bf;
}

// WMMA B-tile (32x16 bf16) from LDS transposed activations lds[n][k], stride ldk.
// Lane L: N = L%16; elements 0..7 -> K = base+0..7, 8..15 -> K = base+16..23, base = (L>=16)*8.
__device__ __forceinline__ v16bf load_b(const unsigned short* __restrict__ L,
                                        int n0, int k0, int ldk, int lane) {
  const unsigned short* p = L + (size_t)(n0 + (lane & 15)) * ldk + k0 + ((lane >> 4) << 3);
  BFVec r;
  r.h[0] = *(const u16x8*)(p);
  r.h[1] = *(const u16x8*)(p + 16);
  return r.bf;
}

__device__ __forceinline__ v8f wmma_bf16(v16bf a, v16bf b, v8f c) {
  return __builtin_amdgcn_wmma_f32_16x16x32_bf16(false, a, false, b, (short)0, c, false, false);
}

// K-loop for one 16x16 D tile: D = W[o0:,:K] * X[:K, n0:]
template <int KTOT>
__device__ __forceinline__ v8f gemm_tile(const unsigned short* __restrict__ W, int o0, int ldw,
                                         const unsigned short* __restrict__ Xl, int n0, int ldx,
                                         int lane) {
  v8f acc = v8f_zero();
#pragma unroll
  for (int k = 0; k < KTOT; k += 32) {
    v16bf a = load_a(W, o0, k, ldw, lane);
    v16bf b = load_b(Xl, n0, k, ldx, lane);
    acc = wmma_bf16(a, b, acc);
  }
  return acc;
}

// relu(D) -> LDS transposed bf16 ([n][k], stride ldk), packed 16B store per lane.
__device__ __forceinline__ void store_d_relu_T(unsigned short* __restrict__ L, v8f d,
                                               int o0, int n0, int ldk, int lane) {
  int n = n0 + (lane & 15);
  int k = o0 + ((lane >> 4) << 3);
  u16x8 pk;
#pragma unroll
  for (int v = 0; v < 8; ++v) pk[v] = f2bf(fmaxf(d[v], 0.f));
  *(u16x8*)(L + (size_t)n * ldk + k) = pk;
}

// ---------------------------------------------------------------- constants
#define BATCH 8
#define NPOS  4096            // 64*64
#define NTILE 128             // columns per workgroup
#define NWG_PER_B 32          // 4096/128
#define VOUT  128
#define VA_IN 256

// workspace element offsets of bf16 weights
#define OFF_WA1 0
#define OFF_WA2 32768
#define OFF_WGB 49152
#define OFF_WGC 81920
#define OFF_WB1 114688
#define OFF_WB2 131072
#define OFF_WC1 147456
#define OFF_WC2 155648
#define OFF_WR1 172032
#define OFF_WR2 270336
#define WB_BYTES   671744                         // 335872 elems * 2
#define OFF_EMBA   ((size_t)WB_BYTES)             // fp32 [8,128,4096]
#define OFF_GATEB  (OFF_EMBA  + 16777216u)
#define OFF_GATEC  (OFF_GATEB + 16777216u)
#define OFF_AGG    (OFF_GATEC + 16777216u)        // bf16 [8,384,4096]

// ---------------------------------------------------------------- K0: fp32 -> bf16
__global__ void k_cvt(const float* __restrict__ src, unsigned short* __restrict__ dst, int n) {
  int i = blockIdx.x * blockDim.x + threadIdx.x;
  if (i < n) dst[i] = f2bf(src[i]);
}

// ---------------------------------------------------------------- K1: emb_a + gates
// per WG: one (batch, 128-col) tile.  X(256x128) in LDS; t1 = relu(Wa1@X) chained
// into emb_a = Wa2@t1; gates from X.
__global__ __launch_bounds__(256) void k_embA(const float* __restrict__ vert_a,
                                              const unsigned short* __restrict__ wa1,
                                              const unsigned short* __restrict__ wa2,
                                              const unsigned short* __restrict__ wgb,
                                              const unsigned short* __restrict__ wgc,
                                              float* __restrict__ emb_a,
                                              float* __restrict__ gate_b,
                                              float* __restrict__ gate_c,
                                              unsigned short* __restrict__ agg) {
  extern __shared__ char dynsm[];
  const int LDX = VA_IN + 8, LDT = VOUT + 8;
  unsigned short* ldsX = (unsigned short*)dynsm;                 // [128][264]
  unsigned short* ldsT = ldsX + 128 * LDX;                       // [128][136]

  int b = blockIdx.x >> 5, nb = blockIdx.x & 31;
  int n0g = nb * NTILE;
  int tid = threadIdx.x, lane = tid & 31, wv = tid >> 5;

  const float* xa = vert_a + (size_t)b * VA_IN * NPOS;
  for (int i = tid; i < VA_IN * (NTILE / 4); i += 256) {          // float4 loads
    int c = i >> 5, n4 = (i & 31) << 2;
    float4 x = *(const float4*)(xa + (size_t)c * NPOS + n0g + n4);
    ldsX[(n4 + 0) * LDX + c] = f2bf(x.x);
    ldsX[(n4 + 1) * LDX + c] = f2bf(x.y);
    ldsX[(n4 + 2) * LDX + c] = f2bf(x.z);
    ldsX[(n4 + 3) * LDX + c] = f2bf(x.w);
  }
  __syncthreads();

  // t1 = relu(Wa1 @ X) -> ldsT (transposed)
  for (int nt = 0; nt < 8; ++nt) {
    v8f d = gemm_tile<VA_IN>(wa1, wv * 16, VA_IN, ldsX, nt * 16, LDX, lane);
    store_d_relu_T(ldsT, d, wv * 16, nt * 16, LDT, lane);
  }
  __syncthreads();

  size_t obase = ((size_t)b * VOUT) * NPOS + n0g;
  size_t abase = ((size_t)b * 384 + 256) * NPOS + n0g;            // emb_a slab of agg
  int row0 = wv * 16 + ((lane >> 4) << 3);
  for (int nt = 0; nt < 8; ++nt) {
    v8f d = gemm_tile<VOUT>(wa2, wv * 16, VOUT, ldsT, nt * 16, LDT, lane);
    int col = nt * 16 + (lane & 15);
#pragma unroll
    for (int v = 0; v < 8; ++v) {
      float x = d[v];
      emb_a[obase + (size_t)(row0 + v) * NPOS + col] = x;
      agg[abase + (size_t)(row0 + v) * NPOS + col] = f2bf(1.f / (1.f + __expf(-x)));
    }
  }
  for (int g = 0; g < 2; ++g) {
    const unsigned short* W = g ? wgc : wgb;
    float* out = g ? gate_c : gate_b;
    for (int nt = 0; nt < 8; ++nt) {
      v8f d = gemm_tile<VA_IN>(W, wv * 16, VA_IN, ldsX, nt * 16, LDX, lane);
      int col = nt * 16 + (lane & 15);
#pragma unroll
      for (int v = 0; v < 8; ++v)
        out[obase + (size_t)(row0 + v) * NPOS + col] = 1.f - 1.f / (1.f + __expf(-d[v]));
    }
  }
}

// ---------------------------------------------------------------- K2/K3: emb_p + attend
// Streams the r^2 patches with an ONLINE SOFTMAX over r; emb_p tiles stay in VGPRs.
template <int R, int CIN>
__global__ __launch_bounds__(256) void k_attend(const float* __restrict__ vert_p,
                                                const unsigned short* __restrict__ w1,  // [128,CIN]
                                                const unsigned short* __restrict__ w2,  // [128,128]
                                                const float* __restrict__ Wrw,          // [256] fp32
                                                const float* __restrict__ emb_a,
                                                const float* __restrict__ gate,
                                                unsigned short* __restrict__ agg,
                                                int slab) {
  constexpr int R2 = R * R;
  constexpr int LDV = CIN + 8;
  const int LDT = VOUT + 8;
  extern __shared__ char dynsm[];
  unsigned short* ldsV = (unsigned short*)dynsm;                  // [128][LDV]
  unsigned short* ldsT = ldsV + 128 * LDV;                        // [128][136]
  float* fls  = (float*)(ldsT + 128 * LDT);
  float* dotA = fls;        float* dotP = fls + 128;
  float* mM   = fls + 256;  float* sS   = fls + 384;
  float* rS   = fls + 512;  float* pP   = fls + 640;

  int b = blockIdx.x >> 5, nb = blockIdx.x & 31;
  int n0g = nb * NTILE;
  int tid = threadIdx.x, lane = tid & 31, wv = tid >> 5;
  const int HR = 64 * R;
  const float* vp = vert_p + (size_t)b * CIN * HR * HR;

  if (tid < 128) { mM[tid] = -1e30f; sS[tid] = 0.f; dotA[tid] = 0.f; }
  __syncthreads();
  {                                                               // dotA[n] = emb_a[:,n] . w1
    int n = tid & 127, half = tid >> 7;
    const float* ea = emb_a + ((size_t)b * VOUT) * NPOS + n0g + n;
    float sm = 0.f;
    for (int c = half * 64; c < half * 64 + 64; ++c) sm += ea[(size_t)c * NPOS] * Wrw[c];
    atomicAdd(&dotA[n], sm);
  }

  float w2r[8];                                                   // W_rw second half, this lane's rows
#pragma unroll
  for (int v = 0; v < 8; ++v) w2r[v] = Wrw[128 + wv * 16 + ((lane >> 4) << 3) + v];

  v8f acc[8];
#pragma unroll
  for (int nt = 0; nt < 8; ++nt) acc[nt] = v8f_zero();

  for (int r = 0; r < R2; ++r) {
    int kh = r / R, kw = r % R;
    __syncthreads();
    for (int i = tid; i < CIN * NTILE; i += 256) {                // unfold gather -> LDS (T)
      int c = i >> 7, n = i & 127;
      int ng = n0g + n, h = ng >> 6, w = ng & 63;
      ldsV[n * LDV + c] = f2bf(vp[((size_t)c * HR + (h * R + kh)) * HR + (w * R + kw)]);
    }
    if (r + 1 < R2) {                                             // prefetch next patch
      int kh2 = (r + 1) / R, kw2 = (r + 1) % R;
      int c = tid & (CIN - 1), ng = n0g + ((tid >> 7) << 6);
      int h = ng >> 6, w = ng & 63;
      __builtin_prefetch(vp + ((size_t)c * HR + (h * R + kh2)) * HR + (w * R + kw2), 0, 1);
    }
    if (tid < 128) dotP[tid] = 0.f;
    __syncthreads();

    for (int nt = 0; nt < 8; ++nt) {                              // t = relu(W1 @ v)
      v8f d = gemm_tile<CIN>(w1, wv * 16, CIN, ldsV, nt * 16, LDV, lane);
      store_d_relu_T(ldsT, d, wv * 16, nt * 16, LDT, lane);
    }
    __syncthreads();

    v8f ep[8];
    for (int nt = 0; nt < 8; ++nt) {                              // emb_p = W2 @ t + logit dot
      ep[nt] = gemm_tile<VOUT>(w2, wv * 16, VOUT, ldsT, nt * 16, LDT, lane);
      float part = 0.f;
#pragma unroll
      for (int v = 0; v < 8; ++v) part += ep[nt][v] * w2r[v];
      atomicAdd(&dotP[nt * 16 + (lane & 15)], part);
    }
    __syncthreads();

    if (tid < 128) {                                              // online softmax state
      float lg = fmaxf(dotA[tid] + dotP[tid], 0.f);
      float mo = mM[tid], mn = fmaxf(lg, mo);
      float rs = __expf(mo - mn), pp = __expf(lg - mn);
      sS[tid] = sS[tid] * rs + pp;
      mM[tid] = mn; rS[tid] = rs; pP[tid] = pp;
    }
    __syncthreads();

#pragma unroll
    for (int nt = 0; nt < 8; ++nt) {                              // rescale + accumulate
      int n = nt * 16 + (lane & 15);
      float rs = rS[n], pp = pP[n];
#pragma unroll
      for (int v = 0; v < 8; ++v) acc[nt][v] = acc[nt][v] * rs + pp * ep[nt][v];
    }
  }

  size_t gbase = ((size_t)b * VOUT) * NPOS + n0g;
  size_t abase = ((size_t)b * 384 + (size_t)slab * VOUT) * NPOS + n0g;
  int row0 = wv * 16 + ((lane >> 4) << 3);
  for (int nt = 0; nt < 8; ++nt) {
    int col = nt * 16 + (lane & 15);
    float inv = 1.f / sS[nt * 16 + (lane & 15)];
#pragma unroll
    for (int v = 0; v < 8; ++v) {
      float g = gate[gbase + (size_t)(row0 + v) * NPOS + col];
      float x = acc[nt][v] * inv * g;
      agg[abase + (size_t)(row0 + v) * NPOS + col] = f2bf(1.f / (1.f + __expf(-x)));
    }
  }
}

// ---------------------------------------------------------------- K4: out = Wr2 @ relu(Wr1 @ agg)
__global__ __launch_bounds__(256) void k_final(const unsigned short* __restrict__ agg,
                                               const unsigned short* __restrict__ wr1, // [256,384]
                                               const unsigned short* __restrict__ wr2, // [256,256]
                                               float* __restrict__ out) {
  const int LDA = 384 + 8, LDY = 256 + 8;
  extern __shared__ char dynsm[];
  unsigned short* ldsA = (unsigned short*)dynsm;                  // [128][392]
  unsigned short* ldsY = ldsA + 128 * LDA;                        // [128][264]

  int b = blockIdx.x >> 5, nb = blockIdx.x & 31;
  int n0g = nb * NTILE;
  int tid = threadIdx.x, lane = tid & 31, wv = tid >> 5;

  const unsigned short* ag = agg + (size_t)b * 384 * NPOS;
  for (int i = tid; i < 384 * (NTILE / 8); i += 256) {            // bf16 already; 16B loads
    int k = i >> 4, n8 = (i & 15) << 3;
    u16x8 v = *(const u16x8*)(ag + (size_t)k * NPOS + n0g + n8);
#pragma unroll
    for (int j = 0; j < 8; ++j) ldsA[(n8 + j) * LDA + k] = v[j];
  }
  __syncthreads();

  for (int ot = 0; ot < 2; ++ot) {                                // y = relu(Wr1 @ agg), M=256
    int o0 = (wv + ot * 8) * 16;
    for (int nt = 0; nt < 8; ++nt) {
      v8f d = gemm_tile<384>(wr1, o0, 384, ldsA, nt * 16, LDA, lane);
      store_d_relu_T(ldsY, d, o0, nt * 16, LDY, lane);
    }
  }
  __syncthreads();

  size_t obase = ((size_t)b * 256) * NPOS + n0g;
  for (int ot = 0; ot < 2; ++ot) {
    int o0 = (wv + ot * 8) * 16;
    for (int nt = 0; nt < 8; ++nt) {
      v8f d = gemm_tile<256>(wr2, o0, 256, ldsY, nt * 16, LDY, lane);
      int row = o0 + ((lane >> 4) << 3), col = nt * 16 + (lane & 15);
#pragma unroll
      for (int v = 0; v < 8; ++v)
        out[obase + (size_t)(row + v) * NPOS + col] = d[v];
    }
  }
}

// ---------------------------------------------------------------- launch
extern "C" void kernel_launch(void* const* d_in, const int* in_sizes, int n_in,
                              void* d_out, int out_size, void* d_ws, size_t ws_size,
                              hipStream_t stream) {
  (void)in_sizes; (void)n_in; (void)out_size; (void)ws_size;
  const float* vert_a = (const float*)d_in[0];
  const float* vert_b = (const float*)d_in[1];
  const float* vert_c = (const float*)d_in[2];
  const float* W_ab   = (const float*)d_in[11];
  const float* W_ac   = (const float*)d_in[12];

  unsigned short* wbf = (unsigned short*)d_ws;
  char* wsb = (char*)d_ws;
  float*          emb_a  = (float*)(wsb + OFF_EMBA);
  float*          gate_b = (float*)(wsb + OFF_GATEB);
  float*          gate_c = (float*)(wsb + OFF_GATEC);
  unsigned short* aggp   = (unsigned short*)(wsb + OFF_AGG);

  // K0: convert all GEMM weights to bf16 in workspace
  struct Cvt { int din; int off; int n; };
  const Cvt cv[10] = {
      {3,  OFF_WA1, 128 * 256}, {4,  OFF_WA2, 128 * 128},
      {5,  OFF_WGB, 128 * 256}, {6,  OFF_WGC, 128 * 256},
      {7,  OFF_WB1, 128 * 128}, {8,  OFF_WB2, 128 * 128},
      {9,  OFF_WC1, 128 * 64},  {10, OFF_WC2, 128 * 128},
      {13, OFF_WR1, 256 * 384}, {14, OFF_WR2, 256 * 256}};
  for (int i = 0; i < 10; ++i)
    k_cvt<<<(cv[i].n + 255) / 256, 256, 0, stream>>>((const float*)d_in[cv[i].din],
                                                     wbf + cv[i].off, cv[i].n);

  const int grid = BATCH * NWG_PER_B;  // 256 WGs, 8 wave32 each

  size_t lds1 = (size_t)(128 * (VA_IN + 8) + 128 * (VOUT + 8)) * 2;
  k_embA<<<grid, 256, lds1, stream>>>(vert_a, wbf + OFF_WA1, wbf + OFF_WA2,
                                      wbf + OFF_WGB, wbf + OFF_WGC,
                                      emb_a, gate_b, gate_c, aggp);

  size_t lds2 = (size_t)(128 * (128 + 8) + 128 * (VOUT + 8)) * 2 + 6 * 128 * 4;
  k_attend<2, 128><<<grid, 256, lds2, stream>>>(vert_b, wbf + OFF_WB1, wbf + OFF_WB2,
                                                W_ab, emb_a, gate_b, aggp, 0);

  size_t lds3 = (size_t)(128 * (64 + 8) + 128 * (VOUT + 8)) * 2 + 6 * 128 * 4;
  k_attend<4, 64><<<grid, 256, lds3, stream>>>(vert_c, wbf + OFF_WC1, wbf + OFF_WC2,
                                               W_ac, emb_a, gate_c, aggp, 1);

  size_t lds4 = (size_t)(128 * (384 + 8) + 128 * (256 + 8)) * 2;
  k_final<<<grid, 256, lds4, stream>>>(aggp, wbf + OFF_WR1, wbf + OFF_WR2, (float*)d_out);
}
